// MultiHeadLatentAttention_11115375362381
// MI455X (gfx1250) — matrix operand order repaired and checked by reading, not compiled, not run on hardware
//
#include <hip/hip_runtime.h>
#include <hip/hip_bf16.h>
#include <math.h>

typedef _Float16 half16 __attribute__((ext_vector_type(16)));
typedef float    floatx8 __attribute__((ext_vector_type(8)));
typedef float    floatx4 __attribute__((ext_vector_type(4)));

union Frag {
    half16  h;
    floatx4 f4[2];
};

#define EMB   2048
#define LAT   128
#define NH    16
#define SEQ   8192
#define BATCH 4
#define SP    512          // S' = SEQ / NH
#define ROWS  (BATCH*SEQ)  // 32768 projection rows
#define KCH   128          // K-chunk staged in LDS
#define LDSA  136          // padded LDS row stride (halves)

// CDNA5 async cache->LDS copy (ASYNCcnt-tracked), bypassing VGPRs
__device__ __forceinline__ void async_ld_b128(unsigned lds, const void* g)
{
    unsigned long long ga = (unsigned long long)(size_t)g;
    asm volatile("global_load_async_to_lds_b128 %0, %1, off"
                 :: "v"(lds), "v"(ga) : "memory");
}
__device__ __forceinline__ void wait_async0()
{
    asm volatile("s_wait_asynccnt 0" ::: "memory");
}

// ---------------------------------------------------------------------------
// Kernel 0: fold position-independent RoPE into Wq/Wk, transpose all
// projection weights to f16 [n][k] (col-major) for contiguous B-fragment loads
// ---------------------------------------------------------------------------
__global__ __launch_bounds__(256) void prep_weights(
    const float* __restrict__ Wq, const float* __restrict__ Wk,
    const float* __restrict__ Wv,
    const float* __restrict__ bq, const float* __restrict__ bk,
    const float* __restrict__ bv,
    _Float16* __restrict__ wqT, _Float16* __restrict__ wkT,
    _Float16* __restrict__ wvT, float* __restrict__ bfold)
{
    int tid = blockIdx.x * blockDim.x + threadIdx.x;   // 3*128*2048 threads
    int mat = tid / (LAT * EMB);
    int rem = tid % (LAT * EMB);
    int n   = rem / EMB;      // output latent column
    int k   = rem % EMB;      // reduction index
    const float* W = (mat == 0) ? Wq : (mat == 1) ? Wk : Wv;
    _Float16* WT   = (mat == 0) ? wqT : (mat == 1) ? wkT : wvT;

    float w;
    int   i = n & 63;
    // freqs = 10000^(-i/64); position-independent rope -> fixed rotation
    float f = __expf(-(float)i * (9.210340371976184f / 64.0f));
    float c = __cosf(f), s = __sinf(f);
    if (mat == 2) {
        w = W[k * LAT + n];
    } else {
        float w1 = W[k * LAT + i];
        float w2 = W[k * LAT + i + 64];
        w = (n < 64) ? (w1 * c - w2 * s) : (w2 * c + w1 * s);
    }
    WT[(size_t)n * EMB + k] = (_Float16)w;

    if (k == 0) {  // fold biases the same way (once per (mat, n))
        const float* bsrc = (mat == 0) ? bq : (mat == 1) ? bk : bv;
        float bb;
        if (mat == 2) bb = bsrc[n];
        else {
            float b1 = bsrc[i], b2 = bsrc[i + 64];
            bb = (n < 64) ? (b1 * c - b2 * s) : (b2 * c + b1 * s);
        }
        bfold[mat * LAT + n] = bb;
    }
}

__global__ __launch_bounds__(256) void prep_wo(
    const float* __restrict__ Wo, _Float16* __restrict__ woT)
{
    int tid = blockIdx.x * blockDim.x + threadIdx.x;   // 2048*2048 threads
    int n = tid / EMB, k = tid % EMB;
    woT[(size_t)n * EMB + k] = (_Float16)Wo[(size_t)k * EMB + n];
}

// ---------------------------------------------------------------------------
// Kernel 1: fused QKV projection (rope pre-folded).  grid=(2048 row-tiles, 3)
// block=256 (8 waves); A slab (16 x 2048 f32) streamed through LDS in
// 16x128 chunks (non-temporal loads: 805MB read exactly once, keep L2 for
// weights + q/k/v intermediates), converted to f16; each wave owns one
// 16-col output tile.  q,k stored per-(b,h) row-major f16; v transposed.
// ---------------------------------------------------------------------------
__global__ __launch_bounds__(256) void qkv_proj(
    const float* __restrict__ Xq, const float* __restrict__ Xk,
    const float* __restrict__ Xv,
    const _Float16* __restrict__ wqT, const _Float16* __restrict__ wkT,
    const _Float16* __restrict__ wvT, const float* __restrict__ bfold,
    _Float16* __restrict__ qa, _Float16* __restrict__ ka,
    _Float16* __restrict__ vT)
{
    __shared__ _Float16 As[16 * LDSA];

    const int mat  = blockIdx.y;
    const float*    X  = (mat == 0) ? Xq  : (mat == 1) ? Xk  : Xv;
    const _Float16* Wt = (mat == 0) ? wqT : (mat == 1) ? wkT : wvT;

    const int row0 = blockIdx.x * 16;
    const int wave = threadIdx.x >> 5;
    const int lane = threadIdx.x & 31;
    const int l15  = lane & 15;
    const int hs   = lane >> 4;
    const int col0 = wave * 16;

    floatx8 acc = {0.f,0.f,0.f,0.f,0.f,0.f,0.f,0.f};

    for (int k0 = 0; k0 < EMB; k0 += KCH) {
        // cooperative coalesced NT load: 16 rows x 128 f32 -> f16 LDS (padded)
        #pragma unroll
        for (int it = 0; it < 2; ++it) {
            int idx4 = threadIdx.x + it * 256;      // 512 float4 per chunk
            int r    = idx4 >> 5;                   // 32 float4 per row
            int c4   = idx4 & 31;
            floatx4 v = __builtin_nontemporal_load(
                (const floatx4*)(X + (size_t)(row0 + r) * EMB + k0 + c4 * 4));
            _Float16* d = &As[r * LDSA + c4 * 4];
            d[0] = (_Float16)v.x; d[1] = (_Float16)v.y;
            d[2] = (_Float16)v.z; d[3] = (_Float16)v.w;
        }
        __syncthreads();

        #pragma unroll
        for (int kk = 0; kk < KCH; kk += 32) {
            Frag a, b;
            const floatx4* ap = (const floatx4*)&As[l15 * LDSA + kk + hs * 16];
            a.f4[0] = ap[0];            // K 0..7 (or 16..23)
            a.f4[1] = ap[1];            // K 8..15 (or 24..31)
            const floatx4* bp = (const floatx4*)(Wt + (size_t)(col0 + l15) * EMB
                                                 + k0 + kk + hs * 16);
            b.f4[0] = bp[0];            // K 0..7 (or 16..23)
            b.f4[1] = bp[1];            // K 8..15 (or 24..31)
            acc = __builtin_amdgcn_wmma_f32_16x16x32_f16(
                      false, a.h, false, b.h, (short)0, acc, false, false);
        }
        __syncthreads();
    }

    // epilogue: rows row0..row0+15 map to h = m (row0 % 16 == 0), s' const
    const int batch = row0 >> 13;              // /8192
    const int sp    = (row0 & (SEQ - 1)) >> 4; // (row % 8192)/16
    const int col   = col0 + l15;
    const float bc  = bfold[mat * LAT + col];

    #pragma unroll
    for (int r = 0; r < 8; ++r) {
        int   m   = r + 8 * hs;                // head index
        float val = acc[r] + bc;
        int   bh  = batch * NH + m;
        if (mat < 2) {
            _Float16* dst = (mat == 0) ? qa : ka;
            dst[((size_t)bh * SP + sp) * LAT + col] = (_Float16)val;
        } else {
            vT[((size_t)bh * LAT + col) * SP + sp] = (_Float16)val;
        }
    }
}

// ---------------------------------------------------------------------------
// Kernel 2: attention.  grid=(8 qtile-groups, 64 bh), block=128 (4 waves).
// Each wave: 16 queries x 512 keys, online softmax, WMMA for Q.K^T and P.V.
// ---------------------------------------------------------------------------
__global__ __launch_bounds__(128) void attention(
    const _Float16* __restrict__ qa, const _Float16* __restrict__ ka,
    const _Float16* __restrict__ vT, const int* __restrict__ mask,
    _Float16* __restrict__ O)
{
    __shared__ _Float16 Ps[4][16 * 40];   // per-wave P relayout scratch (padded)

    const int bh   = blockIdx.y;
    const int wave = threadIdx.x >> 5;
    const int lane = threadIdx.x & 31;
    const int l15  = lane & 15;
    const int hs   = lane >> 4;
    const int q0   = (blockIdx.x * 4 + wave) * 16;

    const _Float16* qb = qa + (size_t)bh * SP * LAT;
    const _Float16* kb = ka + (size_t)bh * SP * LAT;
    const _Float16* vb = vT + (size_t)bh * LAT * SP;

    // load 16x128 Q tile as 4 A-fragments (kept in registers, reused 16x)
    Frag qf[4];
    #pragma unroll
    for (int d = 0; d < 4; ++d) {
        const floatx4* p = (const floatx4*)(qb + (size_t)(q0 + l15) * LAT
                                            + d * 32 + hs * 16);
        qf[d].f4[0] = p[0];
        qf[d].f4[1] = p[1];
    }

    floatx8 acc[8];
    #pragma unroll
    for (int dt = 0; dt < 8; ++dt) acc[dt] = floatx8{0.f,0.f,0.f,0.f,0.f,0.f,0.f,0.f};
    float mrow[8], lrow[8];
    #pragma unroll
    for (int r = 0; r < 8; ++r) { mrow[r] = -INFINITY; lrow[r] = 0.f; }

    const float scale = 0.08838834764831845f;   // 1/sqrt(128)
    _Float16* ps = &Ps[wave][0];

    for (int kc = 0; kc < SP; kc += 32) {
        // ---- scores: two 16x16 tiles (keys kc..+15 and kc+16..+31) ----
        floatx8 s0 = {0.f,0.f,0.f,0.f,0.f,0.f,0.f,0.f};
        floatx8 s1 = {0.f,0.f,0.f,0.f,0.f,0.f,0.f,0.f};
        #pragma unroll
        for (int d = 0; d < 4; ++d) {
            Frag b0, b1;
            const floatx4* p0 = (const floatx4*)(kb + (size_t)(kc + l15) * LAT
                                                 + d * 32 + hs * 16);
            b0.f4[0] = p0[0]; b0.f4[1] = p0[1];
            const floatx4* p1 = (const floatx4*)(kb + (size_t)(kc + 16 + l15) * LAT
                                                 + d * 32 + hs * 16);
            b1.f4[0] = p1[0]; b1.f4[1] = p1[1];
            s0 = __builtin_amdgcn_wmma_f32_16x16x32_f16(
                     false, qf[d].h, false, b0.h, (short)0, s0, false, false);
            s1 = __builtin_amdgcn_wmma_f32_16x16x32_f16(
                     false, qf[d].h, false, b1.h, (short)0, s1, false, false);
        }

        // ---- scale + mask + online softmax (rows live in 16-lane groups) ----
        float pcorr[8];
        #pragma unroll
        for (int r = 0; r < 8; ++r) {
            int qrow = q0 + r + 8 * hs;
            float v0 = s0[r] * scale;
            float v1 = s1[r] * scale;
            if (mask[qrow * SP + kc + l15] == 0)      v0 = -INFINITY;
            if (mask[qrow * SP + kc + 16 + l15] == 0) v1 = -INFINITY;
            float mx = fmaxf(v0, v1);
            #pragma unroll
            for (int o = 1; o < 16; o <<= 1) mx = fmaxf(mx, __shfl_xor(mx, o, 32));
            float mnew = fmaxf(mrow[r], mx);
            float corr = __expf(mrow[r] - mnew);
            mrow[r] = mnew;
            float p0 = __expf(v0 - mnew);
            float p1 = __expf(v1 - mnew);
            s0[r] = p0; s1[r] = p1;
            float rs = p0 + p1;
            #pragma unroll
            for (int o = 1; o < 16; o <<= 1) rs += __shfl_xor(rs, o, 32);
            lrow[r] = lrow[r] * corr + rs;
            pcorr[r] = corr;
        }
        #pragma unroll
        for (int dt = 0; dt < 8; ++dt)
            #pragma unroll
            for (int r = 0; r < 8; ++r) acc[dt][r] *= pcorr[r];

        // ---- relayout P (C layout f32) -> A fragment (f16) via wave LDS ----
        #pragma unroll
        for (int r = 0; r < 8; ++r) {
            int m = r + 8 * hs;
            ps[m * 40 + l15]      = (_Float16)s0[r];
            ps[m * 40 + 16 + l15] = (_Float16)s1[r];
        }
        asm volatile("s_wait_dscnt 0" ::: "memory");  // wave-local RAW in LDS
        Frag pa;
        {
            const floatx4* pp = (const floatx4*)(ps + l15 * 40 + hs * 16);
            pa.f4[0] = pp[0];
            pa.f4[1] = pp[1];
        }

        // ---- P @ V : 8 d-tiles, B from transposed V (contiguous per lane) ----
        #pragma unroll
        for (int dt = 0; dt < 8; ++dt) {
            Frag bv;
            const floatx4* vp = (const floatx4*)(vb + (size_t)(dt * 16 + l15) * SP
                                                 + kc + hs * 16);
            bv.f4[0] = vp[0]; bv.f4[1] = vp[1];
            acc[dt] = __builtin_amdgcn_wmma_f32_16x16x32_f16(
                          false, pa.h, false, bv.h, (short)0, acc[dt], false, false);
        }
    }

    // ---- epilogue: normalize, store f16 in final-GEMM row layout ----
    const int batch = bh >> 4, head = bh & 15;
    #pragma unroll
    for (int dt = 0; dt < 8; ++dt)
        #pragma unroll
        for (int r = 0; r < 8; ++r) {
            int m = r + 8 * hs;
            float o = acc[dt][r] / lrow[r];
            O[((size_t)batch * SP + q0 + m) * EMB + head * LAT + dt * 16 + l15]
                = (_Float16)o;
        }
}

// ---------------------------------------------------------------------------
// Kernel 3: output projection  [2048 x 2048] @ WoT + bo -> f32 d_out
// grid=(128 row-tiles, 16 col-groups), block=256 (8 waves).
// A chunks staged cache->LDS via CDNA5 GLOBAL_LOAD_ASYNC_TO_LDS_B128 with
// manual DOUBLE BUFFERING: chunk i+1's async copy overlaps chunk i's WMMAs;
// ASYNCcnt fenced once per iteration (s_wait_asynccnt 0 + barrier).
// ---------------------------------------------------------------------------
__global__ __launch_bounds__(256) void out_proj(
    const _Float16* __restrict__ O, const _Float16* __restrict__ woT,
    const float* __restrict__ bo, float* __restrict__ out)
{
    __shared__ _Float16 As[2][16 * LDSA];

    const int row0 = blockIdx.x * 16;
    const int wave = threadIdx.x >> 5;
    const int lane = threadIdx.x & 31;
    const int l15  = lane & 15;
    const int hs   = lane >> 4;
    const int col0 = blockIdx.y * 128 + wave * 16;

    // per-thread stage addresses: one b128 (8 halves) per thread per chunk
    const int sr  = threadIdx.x >> 4;
    const int sc8 = threadIdx.x & 15;
    const unsigned ldsA0 = (unsigned)(size_t)&As[0][sr * LDSA + sc8 * 8];
    const unsigned ldsA1 = (unsigned)(size_t)&As[1][sr * LDSA + sc8 * 8];
    const _Float16* gbase = O + (size_t)(row0 + sr) * EMB + sc8 * 8;

    floatx8 acc = {0.f,0.f,0.f,0.f,0.f,0.f,0.f,0.f};

    // prologue: fill buffer 0 with chunk 0
    async_ld_b128(ldsA0, gbase);
    wait_async0();
    __syncthreads();

    for (int k0 = 0; k0 < EMB; k0 += KCH) {
        const int buf = (k0 / KCH) & 1;
        // overlap: start async copy of next chunk into the other buffer
        if (k0 + KCH < EMB)
            async_ld_b128(buf ? ldsA0 : ldsA1, gbase + k0 + KCH);

        #pragma unroll
        for (int kk = 0; kk < KCH; kk += 32) {
            Frag a, b;
            const floatx4* ap = (const floatx4*)&As[buf][l15 * LDSA + kk + hs * 16];
            a.f4[0] = ap[0];
            a.f4[1] = ap[1];
            const floatx4* bp = (const floatx4*)(woT + (size_t)(col0 + l15) * EMB
                                                 + k0 + kk + hs * 16);
            b.f4[0] = bp[0]; b.f4[1] = bp[1];
            acc = __builtin_amdgcn_wmma_f32_16x16x32_f16(
                      false, a.h, false, b.h, (short)0, acc, false, false);
        }
        // next chunk's copy done + all waves' fragment reads of this buffer
        // consumed (compiler-inserted s_wait_dscnt before wmma) -> safe swap
        wait_async0();
        __syncthreads();
    }

    const float bc = bo[col0 + l15];
    #pragma unroll
    for (int r = 0; r < 8; ++r)
        __builtin_nontemporal_store(acc[r] + bc,
            out + (size_t)(row0 + r + 8 * hs) * EMB + col0 + l15);
}

// ---------------------------------------------------------------------------
extern "C" void kernel_launch(void* const* d_in, const int* in_sizes, int n_in,
                              void* d_out, int out_size, void* d_ws, size_t ws_size,
                              hipStream_t stream)
{
    const float* Xq = (const float*)d_in[0];
    const float* Xk = (const float*)d_in[1];
    const float* Xv = (const float*)d_in[2];
    const int*   mk = (const int*)  d_in[3];
    const float* Wq = (const float*)d_in[4];
    const float* bq = (const float*)d_in[5];
    const float* Wk = (const float*)d_in[6];
    const float* bk = (const float*)d_in[7];
    const float* Wv = (const float*)d_in[8];
    const float* bv = (const float*)d_in[9];
    const float* Wo = (const float*)d_in[10];
    const float* bo = (const float*)d_in[11];
    float* out = (float*)d_out;

    char*  ws  = (char*)d_ws;
    size_t off = 0;
    auto take = [&](size_t bytes) -> char* {
        char* p = ws + off;
        off = (off + bytes + 255) & ~(size_t)255;
        return p;
    };
    _Float16* wqT = (_Float16*)take((size_t)LAT * EMB * 2);
    _Float16* wkT = (_Float16*)take((size_t)LAT * EMB * 2);
    _Float16* wvT = (_Float16*)take((size_t)LAT * EMB * 2);
    float*    bfold = (float*) take(3 * LAT * 4);
    _Float16* woT = (_Float16*)take((size_t)EMB * EMB * 2);
    _Float16* qa  = (_Float16*)take((size_t)ROWS * LAT * 2);
    _Float16* ka  = (_Float16*)take((size_t)ROWS * LAT * 2);
    _Float16* vTb = (_Float16*)take((size_t)ROWS * LAT * 2);
    _Float16* Ob  = (_Float16*)take((size_t)BATCH * SP * EMB * 2);

    prep_weights<<<(3 * LAT * EMB) / 256, 256, 0, stream>>>(
        Wq, Wk, Wv, bq, bk, bv, wqT, wkT, wvT, bfold);
    prep_wo<<<(EMB * EMB) / 256, 256, 0, stream>>>(Wo, woT);

    qkv_proj<<<dim3(ROWS / 16, 3), 256, 0, stream>>>(
        Xq, Xk, Xv, wqT, wkT, wvT, bfold, qa, ka, vTb);

    attention<<<dim3(SP / 16 / 4, BATCH * NH), 128, 0, stream>>>(
        qa, ka, vTb, mk, Ob);

    out_proj<<<dim3((BATCH * SP) / 16, EMB / 128), 256, 0, stream>>>(
        Ob, woT, bo, out);
}
